// GENEdge_53412213293641
// MI455X (gfx1250) — matrix operand order repaired and checked by reading, not compiled, not run on hardware
//
#include <hip/hip_runtime.h>

// CDNA5 / gfx1250: wave32, WMMA 16x16x32 bf16 -> f32 accumulate.
typedef __attribute__((ext_vector_type(16))) __bf16 v16bf;
typedef __attribute__((ext_vector_type(8)))  __bf16 v8bf;
typedef __attribute__((ext_vector_type(8)))  float  v8f;

#define WT_LD 40   // bf16 stride of transposed weight chunk in LDS: [128 cols][32 k] (80B = 5*16B)

__device__ inline __bf16 f2bf(float f) { return (__bf16)f; }
__device__ inline float  bf2f(__bf16 b) { return (float)b; }

// Build a 16-half fragment from two 16-byte LDS runs (2x ds_load_b128).
__device__ inline v16bf frag16(const __bf16* p0, const __bf16* p1) {
  v8bf lo = *(const v8bf*)p0;
  v8bf hi = *(const v8bf*)p1;
  return __builtin_shufflevector(lo, hi, 0, 1, 2, 3, 4, 5, 6, 7,
                                 8, 9, 10, 11, 12, 13, 14, 15);
}

// One MLP layer on a 64-row tile: out[64x128] = act(In[64xKp] @ W[Kpx128] + bias)
// 256 threads = 8 waves; wave w computes columns [16w,16w+16) for all 4 row-tiles
// (register blocking: 4 accumulators, one B-frag shared by 4 WMMAs per chunk).
// A-frag (16-bit 16x32): lane L (row=L&15, hi=L>>4) holds K = {hi*8..+8, 16+hi*8..+8}.
// B-frag (32x16): lane holds col=L&15, K = {hi*8..+8, 16+hi*8..+8} (same as A) in the
// transposed [col][k] weight chunk.  Wt is pre-transposed/padded [128][Kp] bf16.
__device__ inline void layer_wmma(const __bf16* __restrict__ Wt,
                                  const float* __restrict__ bias,
                                  int Kp,
                                  const __bf16* __restrict__ actIn, int ldIn,
                                  __bf16* wT,
                                  __bf16* actOut, int ldOut,
                                  __bf16* gOutB,            // bf16 row-major (pre-offset), stride 128
                                  __bf16* gOutT, long ldT,  // bf16 transposed (pre-offset by tile row)
                                  float*  gOutF, int accumOut, // f32 row-major (pre-offset), stride 128
                                  float*  gAtomic, const int* atomRows,
                                  int relu, int tid)
{
  const int lid = tid & 31;
  const int colBase = (tid >> 5) << 4;
  const int r16 = lid & 15;
  const int hi  = lid >> 4;
  v8f acc[4] = {};
  for (int k0 = 0; k0 < Kp; k0 += 32) {
    // stage transposed weight chunk: 128x32 bf16, vectorized 16B copies
    for (int i = tid; i < 128 * 4; i += 256) {
      int cc = i >> 2, k8 = (i & 3) << 3;
      *(v8bf*)(wT + cc * WT_LD + k8) = *(const v8bf*)(Wt + (long)cc * Kp + k0 + k8);
    }
    __syncthreads();
    const __bf16* bp = wT + (colBase + r16) * WT_LD + (hi << 3);
    v16bf b = frag16(bp, bp + 16);
    v16bf a[4];
    const __bf16* ap = actIn + r16 * ldIn + k0 + (hi << 3);
#pragma unroll
    for (int t = 0; t < 4; ++t) {
      const __bf16* p = ap + t * 16 * ldIn;
      a[t] = frag16(p, p + 16);
    }
#pragma unroll
    for (int t = 0; t < 4; ++t)
      acc[t] = __builtin_amdgcn_wmma_f32_16x16x32_bf16(false, a[t], false, b,
                                                       (short)0, acc[t], false, false);
    __syncthreads();
  }
  // C layout: VGPR r -> row = r + 8*hi (+16t), col = colBase + (lane&15)
  const int col = colBase + r16;
  const float bv = bias ? bias[col] : 0.0f;
#pragma unroll
  for (int t = 0; t < 4; ++t) {
    const int rbase = t * 16 + (hi << 3);
    float vals[8];
#pragma unroll
    for (int r = 0; r < 8; ++r) {
      float v = acc[t][r] + bv;
      if (relu) v = fmaxf(v, 0.0f);
      vals[r] = v;
    }
    if (gOutF) {
#pragma unroll
      for (int r = 0; r < 8; ++r) {
        long idx = (long)(rbase + r) * 128 + col;
        float nv = accumOut ? (gOutF[idx] + vals[r]) : vals[r];
        gOutF[idx] = nv;
        vals[r] = nv;
      }
    }
    if (actOut) {
#pragma unroll
      for (int r = 0; r < 8; ++r) actOut[(rbase + r) * ldOut + col] = f2bf(vals[r]);
    }
    if (gOutB) {
#pragma unroll
      for (int r = 0; r < 8; ++r) gOutB[(long)(rbase + r) * 128 + col] = f2bf(vals[r]);
    }
    if (gOutT) {  // packed 16B store: 8 consecutive rows of one column
      v8bf pk;
#pragma unroll
      for (int r = 0; r < 8; ++r) pk[r] = f2bf(vals[r]);
      *(v8bf*)(gOutT + (long)col * ldT + rbase) = pk;
    }
    if (gAtomic) {
#pragma unroll
      for (int r = 0; r < 8; ++r)
        atomicAdd(gAtomic + (long)atomRows[rbase + r] * 128 + col, vals[r]);
    }
  }
  __syncthreads();
}

// weight prep: W[Kx128] f32 -> Wt[128][Kp] bf16 (transposed, zero-padded to Kp)
__global__ void prep_w_kernel(const float* __restrict__ W, __bf16* __restrict__ Wt,
                              int K, int Kp) {
  int i = blockIdx.x * 256 + threadIdx.x;
  if (i < 128 * Kp) {
    int c = i / Kp, k = i % Kp;
    Wt[(long)c * Kp + k] = (k < K) ? f2bf(W[(long)k * 128 + c]) : f2bf(0.0f);
  }
}

// lat [rows][128] f32 -> latT [128][rows] bf16
__global__ void transpose_kernel(const float* __restrict__ src, __bf16* __restrict__ dst,
                                 long rows) {
  long i = (long)blockIdx.x * 256 + threadIdx.x;
  if (i < rows * 128) {
    long r = i >> 7;
    int c = (int)(i & 127);
    dst[(long)c * rows + r] = f2bf(src[i]);
  }
}

__global__ void zero_kernel(float* __restrict__ p, long n) {
  long i = (long)blockIdx.x * blockDim.x + threadIdx.x;
  if (i < n) p[i] = 0.0f;
}

// ---------------- encoder: [x|s] (K=4) -> 128 -> 128 -> 128, writes embT ------
__global__ __launch_bounds__(256) void encoder_kernel(
    const float* __restrict__ x, const float* __restrict__ s,
    const __bf16* Wt0, const float* b0, const __bf16* Wt1, const float* b1,
    const __bf16* Wt2, const float* b2,
    __bf16* __restrict__ embT, long Mtot)
{
  __shared__ __align__(16) __bf16 in32[64 * 40];
  __shared__ __align__(16) __bf16 actA[64 * 136];
  __shared__ __align__(16) __bf16 actB[64 * 136];
  __shared__ __align__(16) __bf16 wT[128 * WT_LD];
  const int tid = threadIdx.x;
  const long m0 = (long)blockIdx.x * 64;
  for (int i = tid; i < 64 * 32; i += 256) {
    int r = i >> 5, c = i & 31;
    long m = m0 + r;
    float v = 0.0f;
    if (c < 3) v = x[m * 3 + c];
    else if (c == 3) v = s[m];
    in32[r * 40 + c] = f2bf(v);
  }
  __syncthreads();
  layer_wmma(Wt0, b0, 32, in32, 40, wT, actA, 136, nullptr, nullptr, 0, nullptr, 0, nullptr, nullptr, 1, tid);
  layer_wmma(Wt1, b1, 128, actA, 136, wT, actB, 136, nullptr, nullptr, 0, nullptr, 0, nullptr, nullptr, 1, tid);
  layer_wmma(Wt2, b2, 128, actB, 136, wT, nullptr, 0, nullptr, embT + m0, Mtot, nullptr, 0, nullptr, nullptr, 0, tid);
}

// --------- streaming GEMM: out[64x128] = A(64xK) @ BT^T, BT = [128][K] bf16 ---
// transA=1: A[k*lda + row] (scores_x columns);  transA=0: A[row*lda + k]
__global__ __launch_bounds__(256) void ab_gemm_kernel(
    const float* __restrict__ A, long lda, int transA,
    const __bf16* __restrict__ BT, long ldBT,
    float* __restrict__ outF, __bf16* __restrict__ outB, int K)
{
  __shared__ __align__(16) __bf16 aStage[64 * 40];
  __shared__ __align__(16) __bf16 wT[128 * WT_LD];
  const int tid = threadIdx.x;
  const long r0 = (long)blockIdx.x * 64;
  const int lid = tid & 31;
  const int colBase = (tid >> 5) << 4, r16 = lid & 15, hi = lid >> 4;
  v8f acc[4] = {};
  for (int k0 = 0; k0 < K; k0 += 32) {
    for (int i = tid; i < 64 * 32; i += 256) {
      int r, c;
      if (transA) { r = i & 63; c = i >> 6; }
      else        { r = i >> 5; c = i & 31; }
      float v = transA ? A[(long)(k0 + c) * lda + r0 + r]
                       : A[(r0 + r) * lda + k0 + c];
      aStage[r * 40 + c] = f2bf(v);
    }
    for (int i = tid; i < 128 * 4; i += 256) {
      int cc = i >> 2, k8 = (i & 3) << 3;
      *(v8bf*)(wT + cc * WT_LD + k8) = *(const v8bf*)(BT + (long)cc * ldBT + k0 + k8);
    }
    if (k0 + 32 < K) {  // prefetch next A chunk (global_prefetch_b8)
      const float* pf = transA
          ? A + (long)(k0 + 32 + (tid >> 6)) * lda + r0 + (tid & 63)
          : A + (r0 + (tid >> 5)) * lda + k0 + 32 + (tid & 31);
      __builtin_prefetch(pf, 0, 1);
    }
    __syncthreads();
    const __bf16* bp = wT + (colBase + r16) * WT_LD + (hi << 3);
    v16bf b = frag16(bp, bp + 16);
    v16bf a[4];
#pragma unroll
    for (int t = 0; t < 4; ++t) {
      const __bf16* p = aStage + (t * 16 + r16) * 40 + (hi << 3);
      a[t] = frag16(p, p + 16);
    }
#pragma unroll
    for (int t = 0; t < 4; ++t)
      acc[t] = __builtin_amdgcn_wmma_f32_16x16x32_bf16(false, a[t], false, b,
                                                       (short)0, acc[t], false, false);
    __syncthreads();
  }
  const int col = colBase + r16;
#pragma unroll
  for (int t = 0; t < 4; ++t) {
    const int rbase = t * 16 + (hi << 3);
#pragma unroll
    for (int r = 0; r < 8; ++r) {
      float v = acc[t][r];
      if (outF) outF[(r0 + rbase + r) * 128 + col] = v;
      if (outB) outB[(r0 + rbase + r) * 128 + col] = f2bf(v);
    }
  }
}

// ------------- edge init: d2 (K=1) -> 128 -> 128 -> 128 -----------------------
__global__ __launch_bounds__(256) void edge_init_kernel(
    const float* __restrict__ pos,
    const int* __restrict__ senders, const int* __restrict__ receivers,
    const __bf16* Wt0, const float* b0, const __bf16* Wt1, const float* b1,
    const __bf16* Wt2, const float* b2, __bf16* __restrict__ edgesB)
{
  __shared__ __align__(16) __bf16 in32[64 * 40];
  __shared__ __align__(16) __bf16 actA[64 * 136];
  __shared__ __align__(16) __bf16 actB[64 * 136];
  __shared__ __align__(16) __bf16 wT[128 * WT_LD];
  const int tid = threadIdx.x;
  const long e0 = (long)blockIdx.x * 64;
  for (int i = tid; i < 64 * 32; i += 256) {
    int r = i >> 5, c = i & 31;
    in32[r * 40 + c] = f2bf(0.0f);
  }
  __syncthreads();
  if (tid < 64) {
    long e = e0 + tid;
    int sd = senders[e], rc = receivers[e];
    float dx = pos[rc * 3 + 0] - pos[sd * 3 + 0];
    float dy = pos[rc * 3 + 1] - pos[sd * 3 + 1];
    float dz = pos[rc * 3 + 2] - pos[sd * 3 + 2];
    in32[tid * 40 + 0] = f2bf(dx * dx + dy * dy + dz * dz);
  }
  __syncthreads();
  layer_wmma(Wt0, b0, 32, in32, 40, wT, actA, 136, nullptr, nullptr, 0, nullptr, 0, nullptr, nullptr, 1, tid);
  layer_wmma(Wt1, b1, 128, actA, 136, wT, actB, 136, nullptr, nullptr, 0, nullptr, 0, nullptr, nullptr, 1, tid);
  layer_wmma(Wt2, b2, 128, actB, 136, wT, nullptr, 0, edgesB + e0 * 128, nullptr, 0, nullptr, 0, nullptr, nullptr, 0, tid);
}

// ----- edge step: [e|lat[recv]|lat[send]] (K=384) -> 128 -> 128 -> 128 --------
// writes edgesB in place + scatter-adds into inbox[recv] (global_atomic_add_f32)
__global__ __launch_bounds__(256) void edge_step_kernel(
    const __bf16* __restrict__ latB, __bf16* __restrict__ edgesB,
    const int* __restrict__ senders, const int* __restrict__ receivers,
    const __bf16* Wt0, const float* b0, const __bf16* Wt1, const float* b1,
    const __bf16* Wt2, const float* b2, float* __restrict__ inbox)
{
  __shared__ __align__(16) __bf16 inBig[64 * 392];
  __shared__ __align__(16) __bf16 actA[64 * 136];
  __shared__ __align__(16) __bf16 actB[64 * 136];
  __shared__ __align__(16) __bf16 wT[128 * WT_LD];
  __shared__ int sidx[64], ridx[64];
  const int tid = threadIdx.x;
  const long e0 = (long)blockIdx.x * 64;
  if (tid < 64) { sidx[tid] = senders[e0 + tid]; ridx[tid] = receivers[e0 + tid]; }
  __syncthreads();
  for (int i = tid; i < 64 * 48; i += 256) {  // 16B vector gather of [e|lat_r|lat_s]
    int r = i / 48, c8 = i % 48;
    int c = c8 << 3;
    const __bf16* src;
    if (c < 128)      src = edgesB + (e0 + r) * 128 + c;
    else if (c < 256) src = latB + (long)ridx[r] * 128 + (c - 128);
    else              src = latB + (long)sidx[r] * 128 + (c - 256);
    *(v8bf*)(inBig + r * 392 + c) = *(const v8bf*)src;
  }
  __syncthreads();
  layer_wmma(Wt0, b0, 384, inBig, 392, wT, actA, 136, nullptr, nullptr, 0, nullptr, 0, nullptr, nullptr, 1, tid);
  layer_wmma(Wt1, b1, 128, actA, 136, wT, actB, 136, nullptr, nullptr, 0, nullptr, 0, nullptr, nullptr, 1, tid);
  layer_wmma(Wt2, b2, 128, actB, 136, wT, nullptr, 0, edgesB + e0 * 128, nullptr, 0, nullptr, 0, inbox, ridx, 0, tid);
}

// ------ node step: [lat|inbox] (K=256) -> 128 -> 128 -> 128, residual add -----
__global__ __launch_bounds__(256) void node_step_kernel(
    float* __restrict__ latF, __bf16* __restrict__ latB,
    const float* __restrict__ inbox,
    const __bf16* Wt0, const float* b0, const __bf16* Wt1, const float* b1,
    const __bf16* Wt2, const float* b2)
{
  __shared__ __align__(16) __bf16 inBig[64 * 264];
  __shared__ __align__(16) __bf16 actA[64 * 136];
  __shared__ __align__(16) __bf16 actB[64 * 136];
  __shared__ __align__(16) __bf16 wT[128 * WT_LD];
  const int tid = threadIdx.x;
  const long n0 = (long)blockIdx.x * 64;
  for (int i = tid; i < 64 * 16; i += 256) {  // lat rows: bf16 16B copies
    int r = i >> 4, c = (i & 15) << 3;
    *(v8bf*)(inBig + r * 264 + c) = *(const v8bf*)(latB + (n0 + r) * 128 + c);
  }
  for (int i = tid; i < 64 * 128; i += 256) { // inbox: f32 -> bf16
    int r = i >> 7, c = i & 127;
    inBig[r * 264 + 128 + c] = f2bf(inbox[(n0 + r) * 128 + c]);
  }
  __syncthreads();
  layer_wmma(Wt0, b0, 256, inBig, 264, wT, actA, 136, nullptr, nullptr, 0, nullptr, 0, nullptr, nullptr, 1, tid);
  layer_wmma(Wt1, b1, 128, actA, 136, wT, actB, 136, nullptr, nullptr, 0, nullptr, 0, nullptr, nullptr, 1, tid);
  layer_wmma(Wt2, b2, 128, actB, 136, wT, nullptr, 0, latB + n0 * 128, nullptr, 0,
             latF + n0 * 128, /*accum=*/1, nullptr, nullptr, 0, tid);
}

// --------- decoder: [z|q] (K=131 -> Kp=160) -> 128 -> 128 -> 1 ----------------
__global__ __launch_bounds__(256) void decoder_kernel(
    const __bf16* __restrict__ zB, const float* __restrict__ q,
    const __bf16* Wt0, const float* b0, const __bf16* Wt1, const float* b1,
    const float* W2, const float* b2, float* __restrict__ out)
{
  __shared__ __align__(16) __bf16 inBig[64 * 168];
  __shared__ __align__(16) __bf16 actA[64 * 136];
  __shared__ __align__(16) __bf16 actB[64 * 136];
  __shared__ __align__(16) __bf16 wT[128 * WT_LD];
  const int tid = threadIdx.x;
  const long q0 = (long)blockIdx.x * 64;
  for (int i = tid; i < 64 * 16; i += 256) {  // z rows: bf16 16B copies
    int r = i >> 4, c = (i & 15) << 3;
    *(v8bf*)(inBig + r * 168 + c) = *(const v8bf*)(zB + (q0 + r) * 128 + c);
  }
  for (int i = tid; i < 64 * 32; i += 256) {  // zero pad cols 128..159
    int r = i >> 5, c = 128 + (i & 31);
    inBig[r * 168 + c] = f2bf(0.0f);
  }
  __syncthreads();
  if (tid < 192) {                            // q cols 128..130
    int r = tid / 3, c = tid % 3;
    inBig[r * 168 + 128 + c] = f2bf(q[(q0 + r) * 3 + c]);
  }
  __syncthreads();
  layer_wmma(Wt0, b0, 160, inBig, 168, wT, actA, 136, nullptr, nullptr, 0, nullptr, 0, nullptr, nullptr, 1, tid);
  layer_wmma(Wt1, b1, 128, actA, 136, wT, actB, 136, nullptr, nullptr, 0, nullptr, 0, nullptr, nullptr, 1, tid);
  // final 128 -> 1: one thread per row
  if (tid < 64) {
    float sum = b2[0];
#pragma unroll 8
    for (int j = 0; j < 128; ++j) sum += bf2f(actB[tid * 136 + j]) * W2[j];
    out[q0 + tid] = sum;
  }
}

extern "C" void kernel_launch(void* const* d_in, const int* in_sizes, int n_in,
                              void* d_out, int out_size, void* d_ws, size_t ws_size,
                              hipStream_t stream)
{
  (void)in_sizes; (void)n_in; (void)out_size; (void)ws_size;
  const int M = 8192, Q = 8192, N = 4096, E = 131072;
  const float* x   = (const float*)d_in[0];
  const float* s   = (const float*)d_in[1];
  const float* q   = (const float*)d_in[2];
  const float* sx  = (const float*)d_in[3];
  const float* sq  = (const float*)d_in[4];
  const float* pos = (const float*)d_in[5];
  // params flattened in jax pytree order (dict keys sorted):
  // blocks[0..7]{edge: W0,b0,W1,b1,W2,b2; node: ...}, decoder(6), edge(6), encoder(6)
  const int P = 6;
  void* const* pp = (void* const*)(d_in + P);
#define BLK_EDGE(i, j) ((const float*)pp[(i) * 12 + (j)])
#define BLK_NODE(i, j) ((const float*)pp[(i) * 12 + 6 + (j)])
#define DEC(j) ((const float*)pp[96 + (j)])
#define EDG(j) ((const float*)pp[102 + (j)])
#define ENC(j) ((const float*)pp[108 + (j)])
  const int* senders   = (const int*)d_in[P + 114];
  const int* receivers = (const int*)d_in[P + 115];

  float* latF  = (float*)d_ws;                 // [N,128] f32 (residual master)
  float* inbox = latF + (long)N * 128;         // [N,128] f32
  __bf16* bptr = (__bf16*)(inbox + (long)N * 128);
  __bf16* embT   = bptr;  bptr += (long)128 * M;  // [128][M]
  __bf16* latB   = bptr;  bptr += (long)N * 128;  // [N,128]
  __bf16* latT   = bptr;  bptr += (long)128 * N;  // [128][N]
  __bf16* edgesB = bptr;  bptr += (long)E * 128;  // [E,128]
  __bf16* zB     = bptr;  bptr += (long)Q * 128;  // [Q,128]
  __bf16* wcur   = bptr;                          // transposed bf16 weights
  float* out = (float*)d_out;

  auto prep = [&](const float* W, int K, int Kp) {
    __bf16* dst = wcur;
    wcur += (long)128 * Kp;
    int n = 128 * Kp;
    prep_w_kernel<<<(n + 255) / 256, 256, 0, stream>>>(W, dst, K, Kp);
    return dst;
  };
  __bf16* wtE0 = prep(ENC(0), 4, 32);
  __bf16* wtE1 = prep(ENC(2), 128, 128);
  __bf16* wtE2 = prep(ENC(4), 128, 128);
  __bf16* wtG0 = prep(EDG(0), 1, 32);
  __bf16* wtG1 = prep(EDG(2), 128, 128);
  __bf16* wtG2 = prep(EDG(4), 128, 128);
  __bf16* wtBE[8][3];
  __bf16* wtBN[8][3];
  for (int st = 0; st < 8; ++st) {
    wtBE[st][0] = prep(BLK_EDGE(st, 0), 384, 384);
    wtBE[st][1] = prep(BLK_EDGE(st, 2), 128, 128);
    wtBE[st][2] = prep(BLK_EDGE(st, 4), 128, 128);
    wtBN[st][0] = prep(BLK_NODE(st, 0), 256, 256);
    wtBN[st][1] = prep(BLK_NODE(st, 2), 128, 128);
    wtBN[st][2] = prep(BLK_NODE(st, 4), 128, 128);
  }
  __bf16* wtD0 = prep(DEC(0), 131, 160);
  __bf16* wtD1 = prep(DEC(2), 128, 128);

  encoder_kernel<<<M / 64, 256, 0, stream>>>(x, s, wtE0, ENC(1), wtE1, ENC(3),
                                             wtE2, ENC(5), embT, (long)M);
  // latents = scores_x^T @ emb   (K = M, A by column, B = embT)
  ab_gemm_kernel<<<N / 64, 256, 0, stream>>>(sx, N, 1, embT, (long)M, latF, latB, M);
  edge_init_kernel<<<E / 64, 256, 0, stream>>>(pos, senders, receivers,
      wtG0, EDG(1), wtG1, EDG(3), wtG2, EDG(5), edgesB);
  for (int st = 0; st < 8; ++st) {
    zero_kernel<<<(N * 128 + 255) / 256, 256, 0, stream>>>(inbox, (long)N * 128);
    edge_step_kernel<<<E / 64, 256, 0, stream>>>(latB, edgesB, senders, receivers,
        wtBE[st][0], BLK_EDGE(st, 1), wtBE[st][1], BLK_EDGE(st, 3),
        wtBE[st][2], BLK_EDGE(st, 5), inbox);
    node_step_kernel<<<N / 64, 256, 0, stream>>>(latF, latB, inbox,
        wtBN[st][0], BLK_NODE(st, 1), wtBN[st][1], BLK_NODE(st, 3),
        wtBN[st][2], BLK_NODE(st, 5));
  }
  // z = scores_q @ latents   (K = N, A by row, B = latT)
  transpose_kernel<<<(N * 128 + 255) / 256, 256, 0, stream>>>(latF, latT, (long)N);
  ab_gemm_kernel<<<Q / 64, 256, 0, stream>>>(sq, N, 0, latT, (long)N, nullptr, zB, N);
  decoder_kernel<<<Q / 64, 256, 0, stream>>>(zB, q, wtD0, DEC(1), wtD1, DEC(3),
                                             DEC(4), DEC(5), out);
#undef BLK_EDGE
#undef BLK_NODE
#undef DEC
#undef EDG
#undef ENC
}
